// MultiHeadAttention_14568529068117
// MI455X (gfx1250) — matrix operand order repaired and checked by reading, not compiled, not run on hardware
//
#include <hip/hip_runtime.h>

#define D_MODEL   1024
#define NUM_HEADS 16
#define HEAD_DIM  64
#define BATCH     2
#define SEQ       2048
#define NEG_BIG   (-1.0e9f)

typedef __attribute__((ext_vector_type(16))) _Float16 v16h;
typedef __attribute__((ext_vector_type(8)))  _Float16 v8h;
typedef __attribute__((ext_vector_type(4)))  _Float16 v4h;
typedef __attribute__((ext_vector_type(8)))  float    v8f;
typedef __attribute__((ext_vector_type(4)))  float    f32x4;

// ---------------------------------------------------------------------------
// CDNA5 async global->LDS copy (ASYNCcnt path, ISA 08_async_tensor GV mode).
// 16B per lane per issue; LDS byte address = low 32 bits of generic pointer.
// ASYNCcnt completes in order -> partial waits give true DMA/compute overlap.
// ---------------------------------------------------------------------------
__device__ __forceinline__ void async_ld_b128(_Float16* lds, const _Float16* g) {
  unsigned int l = (unsigned int)(unsigned long long)(uintptr_t)lds;
  unsigned long long ga = (unsigned long long)(uintptr_t)g;
  asm volatile("global_load_async_to_lds_b128 %0, %1, off"
               :: "v"(l), "v"(ga)
               : "memory");
}
__device__ __forceinline__ void wait_async0() {
  asm volatile("s_wait_asynccnt 0x0" ::: "memory");
}
__device__ __forceinline__ void wait_async4() {  // allow next tile (4) in flight
  asm volatile("s_wait_asynccnt 0x4" ::: "memory");
}

// ---------------------------------------------------------------------------
// WMMA 16x16x32 f16 fragment loaders (CDNA5 wave32 layouts, ISA 7.12.2)
// A (16x32 MxK): lane L row M=L%16; halves 0..7 = K base..base+7,
//                halves 8..15 = K 16+base..16+base+7, base=(L<16?0:8).
// B (32x16 KxN) staged [n][k]: lane L col N=L%16, halves 0..7 = K base..base+7,
//                halves 8..15 = K 16+base..16+base+7, base=(L<16?0:8).
// C/D (16x16 f32): lane L col N=L%16, vgpr r row M = r + (L<16?0:8).
// ---------------------------------------------------------------------------
__device__ __forceinline__ v16h load_frag_a(const _Float16* tile, int ld,
                                            int mbase, int lane) {
  const _Float16* p = tile + (size_t)(mbase + (lane & 15)) * ld + ((lane & 16) ? 8 : 0);
  v8h lo = *(const v8h*)(p);
  v8h hi = *(const v8h*)(p + 16);
  v16h a;
#pragma unroll
  for (int i = 0; i < 8; ++i) { a[i] = lo[i]; a[i + 8] = hi[i]; }
  return a;
}

__device__ __forceinline__ v16h load_frag_b(const _Float16* tile, int ld,
                                            int nbase, int lane) {
  const _Float16* p = tile + (size_t)(nbase + (lane & 15)) * ld + ((lane & 16) ? 8 : 0);
  v8h lo = *(const v8h*)(p);
  v8h hi = *(const v8h*)(p + 16);
  v16h b;
#pragma unroll
  for (int i = 0; i < 8; ++i) { b[i] = lo[i]; b[i + 8] = hi[i]; }
  return b;
}

__device__ __forceinline__ v4h cvt4(f32x4 d) {
  v4h h;
  h[0] = (_Float16)d[0]; h[1] = (_Float16)d[1];
  h[2] = (_Float16)d[2]; h[3] = (_Float16)d[3];
  return h;
}

__device__ __forceinline__ v8f wmma_f16(v16h a, v16h b, v8f c) {
  return __builtin_amdgcn_wmma_f32_16x16x32_f16(false, a, false, b,
                                                (short)0, c, false, false);
}

// ---------------------------------------------------------------------------
// GEMM: Y[M,N] = X[M,K] @ W[K,N] + bias, f16 WMMA inside.
//   TX      : float (raw inputs, VALU-converted staging) or
//             _Float16 (ctx, async global->LDS staging)
//   OUTMODE : 0 = f32 row-major out (final mh_out)
//             1 = f16 row-major out (Q, K intermediates)
//             2 = f16 per-head transposed out Vt[b,h,d,s] (V intermediate)
// 128x128 block tile, 8 waves (2x4), wave tile 64x32 (4x2 frags), BK=32.
// ---------------------------------------------------------------------------
#define GBM 128
#define GBN 128
#define GBK 32
#define GLD 40  // 32 + 8 halves padding (80B rows, 16B aligned)

template <typename TX, int OUTMODE>
__global__ __launch_bounds__(256) void gemm_bias(
    const TX* __restrict__ X, const float* __restrict__ W,
    const float* __restrict__ bias, void* __restrict__ Yv,
    int M, int N, int K) {
  __shared__ __align__(16) _Float16 As[GBM * GLD];
  __shared__ __align__(16) _Float16 Bs[GBN * GLD];

  const int tid  = threadIdx.x;
  const int lane = tid & 31;
  const int wid  = tid >> 5;
  const int wm   = (wid & 1) * 64;
  const int wn   = (wid >> 1) * 32;
  const int m0   = blockIdx.y * GBM;
  const int n0   = blockIdx.x * GBN;

  v8f c[4][2];
#pragma unroll
  for (int mi = 0; mi < 4; ++mi)
#pragma unroll
    for (int ni = 0; ni < 2; ++ni) c[mi][ni] = (v8f)(0.0f);

  for (int k0 = 0; k0 < K; k0 += GBK) {
    // A tile 128x32 -> f16 LDS
    if (sizeof(TX) == 2) {
      // f16 source: async DMA straight into LDS (512 chunks of 8 halves)
#pragma unroll
      for (int i = 0; i < 2; ++i) {
        int ch  = i * 256 + tid;
        int row = ch >> 2;          // 4 chunks per 32-half row
        int col = (ch & 3) << 3;
        async_ld_b128(As + row * GLD + col,
                      (const _Float16*)X + (size_t)(m0 + row) * K + k0 + col);
      }
    } else {
#pragma unroll
      for (int i = 0; i < 4; ++i) {
        int idx = i * 256 + tid;
        int row = idx >> 3;
        int col = (idx & 7) << 2;
        f32x4 d = *(const f32x4*)((const float*)X + (size_t)(m0 + row) * K + k0 + col);
        *(v4h*)(As + row * GLD + col) = cvt4(d);
      }
    }
    // B tile 32x128 of W (f32), stored transposed Bs[n][k]
#pragma unroll
    for (int i = 0; i < 4; ++i) {
      int idx  = i * 256 + tid;
      int krow = idx >> 5;
      int ncol = (idx & 31) << 2;
      f32x4 d = *(const f32x4*)(W + (size_t)(k0 + krow) * N + n0 + ncol);
      Bs[(ncol + 0) * GLD + krow] = (_Float16)d[0];
      Bs[(ncol + 1) * GLD + krow] = (_Float16)d[1];
      Bs[(ncol + 2) * GLD + krow] = (_Float16)d[2];
      Bs[(ncol + 3) * GLD + krow] = (_Float16)d[3];
    }
    if (k0 + GBK < K)
      __builtin_prefetch(W + (size_t)(k0 + GBK) * N + n0 + ((tid & 31) << 2), 0, 1);
    if (sizeof(TX) == 2) wait_async0();
    __syncthreads();

    v16h a[4], b[2];
#pragma unroll
    for (int mi = 0; mi < 4; ++mi) a[mi] = load_frag_a(As, GLD, wm + mi * 16, lane);
#pragma unroll
    for (int ni = 0; ni < 2; ++ni) b[ni] = load_frag_b(Bs, GLD, wn + ni * 16, lane);
#pragma unroll
    for (int mi = 0; mi < 4; ++mi)
#pragma unroll
      for (int ni = 0; ni < 2; ++ni) c[mi][ni] = wmma_f16(a[mi], b[ni], c[mi][ni]);
    __syncthreads();
  }

  const int nlane = lane & 15;
  const int moff  = (lane & 16) ? 8 : 0;
#pragma unroll
  for (int mi = 0; mi < 4; ++mi)
#pragma unroll
    for (int ni = 0; ni < 2; ++ni) {
      int col  = n0 + wn + ni * 16 + nlane;
      float bv = bias[col];
#pragma unroll
      for (int r = 0; r < 8; ++r) {
        int row = m0 + wm + mi * 16 + moff + r;
        float y = c[mi][ni][r] + bv;
        if (OUTMODE == 0) {
          ((float*)Yv)[(size_t)row * N + col] = y;
        } else if (OUTMODE == 1) {
          ((_Float16*)Yv)[(size_t)row * N + col] = (_Float16)y;
        } else {
          // Vt[((b*H + h)*HEAD_DIM + d)*SEQ + s]
          int b_  = row >> 11;          // row / SEQ
          int s_  = row & (SEQ - 1);
          int h_  = col >> 6;           // col / HEAD_DIM
          int d_  = col & (HEAD_DIM - 1);
          ((_Float16*)Yv)[(((size_t)b_ * NUM_HEADS + h_) * HEAD_DIM + d_) * SEQ + s_] =
              (_Float16)y;
        }
      }
    }
}

// ---------------------------------------------------------------------------
// Fused attention staging helpers (async DMA, 4 issues per wave per tile)
// ---------------------------------------------------------------------------
#define SROW 64          // query rows per block
#define SSLD (SEQ + 8)   // 2056 halves (4112B rows, 16B aligned)
#define QLD  72
#define KLD  72
#define VLD  136
#define KBUF 9216        // halves per ping-pong staging buffer (>= 64*VLD)

__device__ __forceinline__ void stage_k_tile(_Float16* buf, const _Float16* Kh,
                                             int b, int h, int n0, int tid) {
#pragma unroll
  for (int i = 0; i < 4; ++i) {
    int ch  = i * 256 + tid;       // 1024 chunks of 8 halves (128x64)
    int row = ch >> 3;
    int col = (ch & 7) << 3;
    async_ld_b128(buf + row * KLD + col,
                  Kh + (size_t)(b * SEQ + n0 + row) * D_MODEL + h * HEAD_DIM + col);
  }
}

__device__ __forceinline__ void stage_v_tile(_Float16* buf, const _Float16* Vt,
                                             int bh, int k0, int tid) {
#pragma unroll
  for (int i = 0; i < 4; ++i) {
    int ch   = i * 256 + tid;      // 1024 chunks of 8 halves (64x128)
    int drow = ch >> 4;
    int col  = (ch & 15) << 3;
    async_ld_b128(buf + drow * VLD + col,
                  Vt + ((size_t)bh * HEAD_DIM + drow) * SEQ + k0 + col);
  }
}

// ---------------------------------------------------------------------------
// Fused attention: per (b,h) and 64-query strip:
//   scores = Qh @ Kh^T * 1/8 + mask*NEG_BIG  -> kept in LDS (f16, 256KB strip)
//   softmax rows in LDS; attn written to HBM exactly once (f32 output)
//   ctx = softmax @ Vh, A-fragments straight from LDS
// K/V tiles stream via double-buffered async DMA (s_wait_asynccnt 4 keeps the
// next tile in flight behind WMMA); first V tile DMA hidden under softmax.
// 256 threads = 8 waves. Dynamic LDS 309,248B (CDNA5 320KB WGP budget).
// ---------------------------------------------------------------------------
__global__ __launch_bounds__(256) void fused_attention(
    const _Float16* __restrict__ Qh, const _Float16* __restrict__ Kh,
    const _Float16* __restrict__ Vt, const float* __restrict__ mask,
    float* __restrict__ attnOut, _Float16* __restrict__ ctx) {
  extern __shared__ __align__(16) _Float16 smem[];
  _Float16* Ss  = smem;                    // SROW x SSLD (131,584 halves)
  _Float16* Qs  = Ss + SROW * SSLD;        // SROW x QLD  (4,608)
  _Float16* St0 = Qs + SROW * QLD;         // ping staging (9,216)
  _Float16* St1 = St0 + KBUF;              // pong staging (9,216)

  const int tid  = threadIdx.x;
  const int lane = tid & 31;
  const int wid  = tid >> 5;
  const int bh   = blockIdx.y;
  const int b    = bh >> 4;
  const int h    = bh & 15;
  const int m0   = blockIdx.x * SROW;

  const float* maskb = mask + (size_t)b * SEQ;

  // ---- stage Q strip: 64 x 64 halves (async DMA, 2 issues/wave) ----
#pragma unroll
  for (int i = 0; i < 2; ++i) {
    int ch  = i * 256 + tid;
    int row = ch >> 3;
    int col = (ch & 7) << 3;
    async_ld_b128(Qs + row * QLD + col,
                  Qh + (size_t)(b * SEQ + m0 + row) * D_MODEL + h * HEAD_DIM + col);
  }

  // ---- phase 1: scores into Ss (f16), double-buffered K tiles ----
  const int wmS = (wid & 1) * 32;    // 2 wave-rows over 64
  const int wnS = (wid >> 1) * 32;   // 4 wave-cols over 128
  const float scale = 0.125f;        // 1/sqrt(64)
  const int nlane = lane & 15;
  const int moff  = (lane & 16) ? 8 : 0;
  const int NT = SEQ / 128;          // 16 tiles

  stage_k_tile(St0, Kh, b, h, 0, tid);
  for (int nt = 0; nt < NT; ++nt) {
    _Float16* cur = (nt & 1) ? St1 : St0;
    if (nt + 1 < NT) {
      stage_k_tile((nt & 1) ? St0 : St1, Kh, b, h, (nt + 1) * 128, tid);
      wait_async4();   // tile nt done; tile nt+1 stays in flight
    } else {
      wait_async0();
    }
    __syncthreads();

    v8f c2[2][2];
#pragma unroll
    for (int mi = 0; mi < 2; ++mi)
#pragma unroll
      for (int ni = 0; ni < 2; ++ni) c2[mi][ni] = (v8f)(0.0f);

#pragma unroll
    for (int kk = 0; kk < HEAD_DIM; kk += 32) {
      v16h a[2], bf[2];
#pragma unroll
      for (int mi = 0; mi < 2; ++mi) a[mi] = load_frag_a(Qs + kk, QLD, wmS + mi * 16, lane);
#pragma unroll
      for (int ni = 0; ni < 2; ++ni) bf[ni] = load_frag_b(cur + kk, KLD, wnS + ni * 16, lane);
#pragma unroll
      for (int mi = 0; mi < 2; ++mi)
#pragma unroll
        for (int ni = 0; ni < 2; ++ni) c2[mi][ni] = wmma_f16(a[mi], bf[ni], c2[mi][ni]);
    }

    int n0 = nt * 128;
#pragma unroll
    for (int mi = 0; mi < 2; ++mi)
#pragma unroll
      for (int ni = 0; ni < 2; ++ni) {
        int col  = wnS + ni * 16 + nlane;
        float mb = maskb[n0 + col] * NEG_BIG;
#pragma unroll
        for (int r = 0; r < 8; ++r) {
          int row = wmS + mi * 16 + moff + r;
          Ss[(size_t)row * SSLD + n0 + col] = (_Float16)(c2[mi][ni][r] * scale + mb);
        }
      }
    __syncthreads();
  }

  // ---- issue first V tile now: DMA hides under the whole softmax phase ----
  stage_v_tile(St0, Vt, bh, 0, tid);

  // ---- phase 2: softmax rows in LDS; write f32 attn to HBM once ----
  // wave w owns rows w*8..w*8+7; lane covers 64 contiguous cols.
  for (int i = 0; i < 8; ++i) {
    int row = wid * 8 + i;
    const _Float16* pr = Ss + (size_t)row * SSLD + lane * 64;
    float ev[64];
    float m = -3.0e38f;
#pragma unroll
    for (int j8 = 0; j8 < 8; ++j8) {
      v8h d = *(const v8h*)(pr + j8 * 8);
#pragma unroll
      for (int e = 0; e < 8; ++e) {
        float v = (float)d[e];
        ev[j8 * 8 + e] = v;
        m = fmaxf(m, v);
      }
    }
#pragma unroll
    for (int off = 16; off > 0; off >>= 1) m = fmaxf(m, __shfl_xor(m, off));

    float sum = 0.0f;
#pragma unroll
    for (int j = 0; j < 64; ++j) {
      ev[j] = __expf(ev[j] - m);
      sum += ev[j];
    }
#pragma unroll
    for (int off = 16; off > 0; off >>= 1) sum += __shfl_xor(sum, off);
    float inv = 1.0f / sum;

    float* po = attnOut + ((size_t)bh * SEQ + m0 + row) * SEQ + lane * 64;
    _Float16* pw = Ss + (size_t)row * SSLD + lane * 64;
#pragma unroll
    for (int j4 = 0; j4 < 16; ++j4) {
      f32x4 o;
      o[0] = ev[j4 * 4 + 0] * inv;
      o[1] = ev[j4 * 4 + 1] * inv;
      o[2] = ev[j4 * 4 + 2] * inv;
      o[3] = ev[j4 * 4 + 3] * inv;
      *(f32x4*)(po + j4 * 4) = o;
      *(v4h*)(pw + j4 * 4) = cvt4(o);
    }
  }
  __syncthreads();

  // ---- phase 3: ctx = attn(LDS f16) @ Vh, double-buffered V tiles ----
  const int wmV = (wid & 3) * 16;    // 4 wave-rows over 64
  const int wnV = (wid >> 2) * 32;   // 2 wave-cols over 64
  v8f cc[2];
  cc[0] = (v8f)(0.0f);
  cc[1] = (v8f)(0.0f);

  for (int t = 0; t < NT; ++t) {
    _Float16* cur = (t & 1) ? St1 : St0;
    if (t + 1 < NT) {
      stage_v_tile((t & 1) ? St0 : St1, Vt, bh, (t + 1) * 128, tid);
      wait_async4();
    } else {
      wait_async0();
    }
    __syncthreads();

    int k0 = t * 128;
#pragma unroll
    for (int kk = 0; kk < 128; kk += 32) {
      v16h a  = load_frag_a(Ss + k0 + kk, SSLD, wmV, lane);
      v16h b0 = load_frag_b(cur + kk, VLD, wnV, lane);
      v16h b1 = load_frag_b(cur + kk, VLD, wnV + 16, lane);
      cc[0] = wmma_f16(a, b0, cc[0]);
      cc[1] = wmma_f16(a, b1, cc[1]);
    }
    __syncthreads();
  }

#pragma unroll
  for (int ni = 0; ni < 2; ++ni) {
    int col = wnV + ni * 16 + nlane;
#pragma unroll
    for (int r = 0; r < 8; ++r) {
      int row = m0 + wmV + moff + r;
      ctx[((size_t)bh * SEQ + row) * HEAD_DIM + col] = (_Float16)cc[ni][r];
    }
  }
}

// ---------------------------------------------------------------------------
extern "C" void kernel_launch(void* const* d_in, const int* in_sizes, int n_in,
                              void* d_out, int out_size, void* d_ws, size_t ws_size,
                              hipStream_t stream) {
  (void)in_sizes; (void)n_in; (void)out_size; (void)ws_size;

  const float* query = (const float*)d_in[0];
  const float* key   = (const float*)d_in[1];
  const float* value = (const float*)d_in[2];
  const float* mask  = (const float*)d_in[3];
  const float* Wq    = (const float*)d_in[4];
  const float* bq    = (const float*)d_in[5];
  const float* Wk    = (const float*)d_in[6];
  const float* bk    = (const float*)d_in[7];
  const float* Wv    = (const float*)d_in[8];
  const float* bv    = (const float*)d_in[9];
  const float* Wo    = (const float*)d_in[10];
  const float* bo    = (const float*)d_in[11];

  const size_t bsd = (size_t)BATCH * SEQ * D_MODEL;  // 4,194,304 elems
  float* mh   = (float*)d_out;                        // [B,S,D] f32
  float* attn = (float*)d_out + bsd;                  // [B,H,S,S] f32

  _Float16* q16 = (_Float16*)d_ws;     // [B,S,D] f16
  _Float16* k16 = q16 + bsd;           // [B,S,D] f16
  _Float16* vt  = k16 + bsd;           // [B,H,dh,S] f16 (per-head transposed)
  _Float16* ctx = vt + bsd;            // [B,H,S,dh] f16 == reference's flat view

  const int M = BATCH * SEQ;  // 4096
  dim3 blk(256);
  dim3 gproj(D_MODEL / GBN, M / GBM);            // (8, 32)
  dim3 gfa(SEQ / SROW, BATCH * NUM_HEADS);       // (32, 32)
  const size_t smem = (size_t)(SROW * SSLD + SROW * QLD + 2 * KBUF) * sizeof(_Float16);

  gemm_bias<float, 1><<<gproj, blk, 0, stream>>>(query, Wq, bq, q16, M, D_MODEL, D_MODEL);
  gemm_bias<float, 1><<<gproj, blk, 0, stream>>>(key,   Wk, bk, k16, M, D_MODEL, D_MODEL);
  gemm_bias<float, 2><<<gproj, blk, 0, stream>>>(value, Wv, bv, vt,  M, D_MODEL, D_MODEL);
  fused_attention<<<gfa, blk, smem, stream>>>(q16, k16, vt, mask, attn, ctx);
  gemm_bias<_Float16, 0><<<gproj, blk, 0, stream>>>(ctx, Wo, bo, mh, M, D_MODEL, D_MODEL);
}
